// GCNTox21_33895881900361
// MI455X (gfx1250) — matrix operand order repaired and checked by reading, not compiled, not run on hardware
//
#include <hip/hip_runtime.h>
#include <hip/hip_bf16.h>
#include <math.h>

typedef __attribute__((ext_vector_type(16))) _Float16 v16h;
typedef __attribute__((ext_vector_type(8)))  _Float16 v8h;
typedef __attribute__((ext_vector_type(8)))  float    v8f;

#define N_GRAPHS 4096

// ---------------------------------------------------------------------------
// Utility: zero a float buffer
// ---------------------------------------------------------------------------
__global__ void zero_f32(float* __restrict__ p, int n) {
    int i = blockIdx.x * blockDim.x + threadIdx.x;
    if (i < n) p[i] = 0.0f;
}

// ---------------------------------------------------------------------------
// Pack an f32 weight matrix W[K x N] (row-major) into WMMA B-fragment order
// (f16).  Fragment layout (16x16x32 f16 B, 32xN per frag):
//   frag index = kc*NT + nt ; within frag: lane in [0,32), half i in [0,16)
//   k_local = 16*(lane>>4) + i ;  n = nt*16 + (lane&15)
// If CIN>0 the K axis is the concat [x_dst(CIN), x_src(CIN), e(16 pad 32)].
// ---------------------------------------------------------------------------
__global__ void pack_frags(const float* __restrict__ W, _Float16* __restrict__ out,
                           int KC, int NT, int N, int CIN) {
    int idx = blockIdx.x * blockDim.x + threadIdx.x;
    int total = KC * NT * 512;
    if (idx >= total) return;
    int frag = idx >> 9;
    int r    = idx & 511;
    int lane = r >> 4;
    int i    = r & 15;
    int kc = frag / NT;
    int nt = frag - kc * NT;
    int kLocal = ((lane >> 4) << 4) + i;
    int n = nt * 16 + (lane & 15);
    int ksrc;
    if (CIN > 0) {
        int kca = CIN / 32;
        if (kc < kca)            ksrc = kc * 32 + kLocal;               // x_dst block
        else if (kc < 2 * kca)   ksrc = CIN + (kc - kca) * 32 + kLocal; // x_src block
        else                     ksrc = (kLocal < 16) ? (2 * CIN + kLocal) : -1; // edge pad
    } else {
        ksrc = kc * 32 + kLocal;
    }
    float v = (ksrc >= 0) ? W[(size_t)ksrc * N + n] : 0.0f;
    out[idx] = (_Float16)v;
}

// ---------------------------------------------------------------------------
// Node embedding: h = relu(x @ Wn + bn), x:[N,32], Wn:[32,64] -> f16 [N,64]
// ---------------------------------------------------------------------------
__global__ void node_embed(const float* __restrict__ x, const float* __restrict__ Wn,
                           const float* __restrict__ bn, _Float16* __restrict__ hOut,
                           int nNodes) {
    int node = blockIdx.x;
    int c = threadIdx.x;
    if (node >= nNodes) return;
    const float* xr = x + (size_t)node * 32;
    float acc = bn[c];
#pragma unroll
    for (int k = 0; k < 32; ++k) acc += xr[k] * Wn[k * 64 + c];
    hOut[(size_t)node * 64 + c] = (_Float16)(acc > 0.0f ? acc : 0.0f);
}

// ---------------------------------------------------------------------------
// Edge embedding: e = relu(edge_attr @ We + be), [E,8]x[8,16] -> f16 [E,16]
// ---------------------------------------------------------------------------
__global__ void edge_embed(const float* __restrict__ ea, const float* __restrict__ We,
                           const float* __restrict__ be, _Float16* __restrict__ eOut,
                           int nEdges) {
    int idx = blockIdx.x * blockDim.x + threadIdx.x;
    if (idx >= nEdges * 16) return;
    int e = idx >> 4;
    int c = idx & 15;
    const float* r = ea + (size_t)e * 8;
    float acc = be[c];
#pragma unroll
    for (int k = 0; k < 8; ++k) acc += r[k] * We[k * 16 + c];
    eOut[idx] = (_Float16)(acc > 0.0f ? acc : 0.0f);
}

// ---------------------------------------------------------------------------
// In-degree counts (mean aggregation denominator) and per-graph node counts
// ---------------------------------------------------------------------------
__global__ void edge_counts(const int* __restrict__ dst, float* __restrict__ cntN, int nEdges) {
    int i = blockIdx.x * blockDim.x + threadIdx.x;
    if (i < nEdges) atomicAdd(&cntN[dst[i]], 1.0f);
}
__global__ void node_counts(const int* __restrict__ batch, float* __restrict__ cntG, int nNodes) {
    int i = blockIdx.x * blockDim.x + threadIdx.x;
    if (i < nNodes) atomicAdd(&cntG[batch[i]], 1.0f);
}

// ---------------------------------------------------------------------------
// Edge-conditioned conv layer.  Each wave processes TPW 16-edge tiles:
//   m = relu([h[dst], h[src], e] @ Wa + ba) @ Wb + bb ;  nodeSum[dst] += m
// Packed weight fragments are staged into (dynamic) LDS once per block via the
// CDNA5 async global->LDS path (ASYNCcnt), so all WMMA B operands come from
// ds_load_b128 and the global/L2 path is reserved for the irregular gathers
// and the atomic scatter.
// ---------------------------------------------------------------------------
template <int CIN, int H, int OUT>
__global__ __launch_bounds__(128) void conv_kernel(
    const _Float16* __restrict__ hIn,   // [N, CIN] f16
    const _Float16* __restrict__ eEmb,  // [E, 16]  f16
    const int* __restrict__ srcIdx, const int* __restrict__ dstIdx,
    const _Float16* __restrict__ WaP, const float* __restrict__ ba,
    const _Float16* __restrict__ WbP, const float* __restrict__ bb,
    float* __restrict__ nodeSum,        // [N, OUT] f32 (pre-zeroed)
    int nEdges) {
    constexpr int KCA = CIN / 32;       // K-chunks per node-feature block
    constexpr int KC1 = 2 * KCA + 1;    // + 1 padded edge chunk
    constexpr int NT1 = H / 16;
    constexpr int KC2 = H / 32;
    constexpr int NT2 = OUT / 16;
    constexpr int WA_HALVES = KC1 * NT1 * 512;
    constexpr int WB_HALVES = KC2 * NT2 * 512;
    constexpr int TPW = 4;              // tiles per wave (amortizes staging)

    extern __shared__ __align__(16) _Float16 smem[];
    _Float16* sWa  = smem;
    _Float16* sWb  = smem + WA_HALVES;
    _Float16* sHidBase = sWb + WB_HALVES;        // 4 waves x 16 x H halves

    // ---- stage packed weight fragments into LDS via async global->LDS ----
    {
        const unsigned ldsA = (unsigned)(size_t)(void*)sWa;   // low 32 bits = LDS offset
        const unsigned ldsB = (unsigned)(size_t)(void*)sWb;
        const unsigned long long gA = (unsigned long long)(size_t)WaP;
        const unsigned long long gB = (unsigned long long)(size_t)WbP;
        const int na = WA_HALVES / 8;                         // 16B chunks
        const int nb = WB_HALVES / 8;
        for (int i = threadIdx.x; i < na; i += blockDim.x) {
            unsigned l = ldsA + (unsigned)i * 16u;
            unsigned long long g = gA + (unsigned long long)i * 16ull;
            asm volatile("global_load_async_to_lds_b128 %0, %1, off"
                         :: "v"(l), "v"(g) : "memory");
        }
        for (int i = threadIdx.x; i < nb; i += blockDim.x) {
            unsigned l = ldsB + (unsigned)i * 16u;
            unsigned long long g = gB + (unsigned long long)i * 16ull;
            asm volatile("global_load_async_to_lds_b128 %0, %1, off"
                         :: "v"(l), "v"(g) : "memory");
        }
        asm volatile("s_wait_asynccnt 0" ::: "memory");
    }
    __syncthreads();

    const int lane = threadIdx.x & 31;
    const int wave = threadIdx.x >> 5;
    const int row = lane & 15;   // edge row within tile (A-matrix M)
    const int hh  = lane >> 4;   // lane half selects K sub-ranges
    const int nLo = lane & 15;
    _Float16* hid = sHidBase + wave * 16 * H;

    // tile-invariant biases, hoisted out of the tile loop
    float baL[NT1], bbL[NT2];
#pragma unroll
    for (int nt = 0; nt < NT1; ++nt) baL[nt] = ba[nt * 16 + nLo];
#pragma unroll
    for (int nt = 0; nt < NT2; ++nt) bbL[nt] = bb[nt * 16 + nLo];

    const int tileBase = (blockIdx.x * 4 + wave) * TPW;
    union HV { v16h v; v8h h[2]; };

    for (int t = 0; t < TPW; ++t) {
        const int tile = tileBase + t;
        if (tile * 16 >= nEdges) break;              // whole-wave exit
        const int eBase = tile * 16;
        const int e0 = eBase + row;
        const int dn = dstIdx[e0];
        const int sn = srcIdx[e0];
        const _Float16* rowD = hIn + (size_t)dn * CIN;
        const _Float16* rowS = hIn + (size_t)sn * CIN;
        const _Float16* rowE = eEmb + (size_t)e0 * 16;

        // ---- GEMM 1: [16 x K] x [K x H] ----
        v8f acc[NT1] = {};
        for (int kc = 0; kc < KC1; ++kc) {
            HV a;
            if (kc < KCA) {
                const _Float16* p = rowD + kc * 32;
                a.h[0] = *(const v8h*)(p + 16 * hh);         // K = 16*hh .. +7
                a.h[1] = *(const v8h*)(p + 16 * hh + 8);     // K = 16*hh+8 .. +7
            } else if (kc < 2 * KCA) {
                const _Float16* p = rowS + (kc - KCA) * 32;
                a.h[0] = *(const v8h*)(p + 16 * hh);
                a.h[1] = *(const v8h*)(p + 16 * hh + 8);
            } else {
                v8h z = {};
                a.h[0] = hh ? z : *(const v8h*)(rowE);       // 16 real K, rest zero
                a.h[1] = hh ? z : *(const v8h*)(rowE + 8);
            }
            for (int nt = 0; nt < NT1; ++nt) {
                const _Float16* bp = sWa + (kc * NT1 + nt) * 512 + lane * 16;
                HV b;
                b.h[0] = *(const v8h*)bp;                    // ds_load_b128
                b.h[1] = *(const v8h*)(bp + 8);
                acc[nt] = __builtin_amdgcn_wmma_f32_16x16x32_f16(
                    false, a.v, false, b.v, (short)0, acc[nt], false, false);
            }
        }

        // ---- bias + ReLU, relayout C->A via LDS (same-wave DS is in-order) ----
        for (int nt = 0; nt < NT1; ++nt) {
            const int n = nt * 16 + nLo;
            const float bias = baL[nt];
#pragma unroll
            for (int r = 0; r < 8; ++r) {
                const int m = r + 8 * hh;                    // C layout: M = r + 8*half
                float v = acc[nt][r] + bias;
                hid[m * H + n] = (_Float16)(v > 0.0f ? v : 0.0f);
            }
        }

        // ---- GEMM 2: [16 x H] x [H x OUT] ----
        v8f acc2[NT2] = {};
        for (int kc = 0; kc < KC2; ++kc) {
            HV a;
            const _Float16* p = hid + row * H + kc * 32;
            a.h[0] = *(const v8h*)(p + 16 * hh);
            a.h[1] = *(const v8h*)(p + 16 * hh + 8);
            for (int nt = 0; nt < NT2; ++nt) {
                const _Float16* bp = sWb + (kc * NT2 + nt) * 512 + lane * 16;
                HV b;
                b.h[0] = *(const v8h*)bp;
                b.h[1] = *(const v8h*)(bp + 8);
                acc2[nt] = __builtin_amdgcn_wmma_f32_16x16x32_f16(
                    false, a.v, false, b.v, (short)0, acc2[nt], false, false);
            }
        }

        // ---- bias + scatter-add messages into destination nodes ----
        int dRows[8];
#pragma unroll
        for (int r = 0; r < 8; ++r) dRows[r] = dstIdx[eBase + r + 8 * hh];
        for (int nt = 0; nt < NT2; ++nt) {
            const int n = nt * 16 + nLo;
            const float bias = bbL[nt];
#pragma unroll
            for (int r = 0; r < 8; ++r)
                atomicAdd(&nodeSum[(size_t)dRows[r] * OUT + n], acc2[nt][r] + bias);
        }
    }
}

// ---------------------------------------------------------------------------
// Mean-aggregate (sum/cnt in place) + per-channel sum & sumsq for batchnorm
// ---------------------------------------------------------------------------
template <int C>
__global__ void finalize_stats(float* __restrict__ nodeSum, const float* __restrict__ cntN,
                               float* __restrict__ stats, int nNodes) {
    __shared__ float ls[2 * C];
    for (int i = threadIdx.x; i < 2 * C; i += blockDim.x) ls[i] = 0.0f;
    __syncthreads();
    const int total = nNodes * C;
    for (int idx = blockIdx.x * blockDim.x + threadIdx.x; idx < total;
         idx += gridDim.x * blockDim.x) {
        int node = idx / C;
        int c = idx - node * C;
        float m = nodeSum[idx] / fmaxf(cntN[node], 1.0f);
        nodeSum[idx] = m;
        atomicAdd(&ls[c], m);
        atomicAdd(&ls[C + c], m * m);
    }
    __syncthreads();
    for (int i = threadIdx.x; i < 2 * C; i += blockDim.x) atomicAdd(&stats[i], ls[i]);
}

// params[c]=scale, params[C+c]=shift
__global__ void bn_params(const float* __restrict__ stats, const float* __restrict__ g,
                          const float* __restrict__ beta, float* __restrict__ params,
                          int C, float invN) {
    int c = threadIdx.x;
    if (c >= C) return;
    float mean = stats[c] * invN;
    float var = stats[C + c] * invN - mean * mean;
    float scale = g[c] * rsqrtf(var + 1e-5f);
    params[c] = scale;
    params[C + c] = beta[c] - mean * scale;
}

// y = relu(bn(x)); write f16 for the next conv; last layer also pools per graph
template <int C, bool POOL>
__global__ void apply_bn(const float* __restrict__ nodeMean, const float* __restrict__ params,
                         _Float16* __restrict__ hOut, const int* __restrict__ batch,
                         float* __restrict__ graphSum, int nNodes) {
    int idx = blockIdx.x * blockDim.x + threadIdx.x;
    if (idx >= nNodes * C) return;
    int node = idx / C;
    int c = idx - node * C;
    float v = nodeMean[idx] * params[c] + params[C + c];
    v = v > 0.0f ? v : 0.0f;
    hOut[idx] = (_Float16)v;
    if (POOL) atomicAdd(&graphSum[(size_t)batch[node] * 16 + c], v);
}

// ---------------------------------------------------------------------------
// Head: sigmoid(meanpool @ Wfc + bfc)  -> out [G,12]
// ---------------------------------------------------------------------------
__global__ void head(const float* __restrict__ graphSum, const float* __restrict__ cntG,
                     const float* __restrict__ Wfc, const float* __restrict__ bfc,
                     float* __restrict__ out, int nGraphs) {
    int idx = blockIdx.x * blockDim.x + threadIdx.x;
    if (idx >= nGraphs * 12) return;
    int g = idx / 12;
    int j = idx - g * 12;
    float inv = 1.0f / fmaxf(cntG[g], 1.0f);
    float acc = bfc[j];
#pragma unroll
    for (int c = 0; c < 16; ++c) acc += graphSum[g * 16 + c] * inv * Wfc[c * 12 + j];
    out[idx] = 1.0f / (1.0f + __expf(-acc));
}

// ---------------------------------------------------------------------------
extern "C" void kernel_launch(void* const* d_in, const int* in_sizes, int n_in,
                              void* d_out, int out_size, void* d_ws, size_t ws_size,
                              hipStream_t stream) {
    const float* x         = (const float*)d_in[0];
    const float* edge_attr = (const float*)d_in[1];
    const int*   edge_idx  = (const int*)d_in[2];
    const int*   batch     = (const int*)d_in[3];
    const float* Wn  = (const float*)d_in[5];  const float* bn_ = (const float*)d_in[6];
    const float* We  = (const float*)d_in[7];  const float* be_ = (const float*)d_in[8];
    const float* W1a = (const float*)d_in[9];  const float* b1a = (const float*)d_in[10];
    const float* W1b = (const float*)d_in[11]; const float* b1b = (const float*)d_in[12];
    const float* W2a = (const float*)d_in[13]; const float* b2a = (const float*)d_in[14];
    const float* W2b = (const float*)d_in[15]; const float* b2b = (const float*)d_in[16];
    const float* W3a = (const float*)d_in[17]; const float* b3a = (const float*)d_in[18];
    const float* W3b = (const float*)d_in[19]; const float* b3b = (const float*)d_in[20];
    const float* g1  = (const float*)d_in[21]; const float* be1 = (const float*)d_in[22];
    const float* g2  = (const float*)d_in[23]; const float* be2 = (const float*)d_in[24];
    const float* g3  = (const float*)d_in[25]; const float* be3 = (const float*)d_in[26];
    const float* Wfc = (const float*)d_in[27]; const float* bfc = (const float*)d_in[28];

    const int N = in_sizes[0] / 32;
    const int E = in_sizes[2] / 2;
    const int G = N_GRAPHS;
    const int* srcIdx = edge_idx;
    const int* dstIdx = edge_idx + E;

    // ---- workspace carve-out ----
    char* ws = (char*)d_ws;
    size_t off = 0;
    auto carve = [&](size_t bytes) -> void* {
        void* p = ws + off;
        off = (off + bytes + 255) & ~(size_t)255;
        return p;
    };
    _Float16* hF16     = (_Float16*)carve((size_t)N * 64 * 2);
    _Float16* eF16     = (_Float16*)carve((size_t)E * 16 * 2);
    float*    nodeSum  = (float*)carve((size_t)N * 64 * 4);
    float*    cntN     = (float*)carve((size_t)N * 4);
    float*    cntG     = (float*)carve((size_t)G * 4);
    float*    graphSum = (float*)carve((size_t)G * 16 * 4);
    float*    stats    = (float*)carve(128 * 4);
    float*    params   = (float*)carve(128 * 4);
    _Float16* pWa1 = (_Float16*)carve((size_t)5 * 8 * 512 * 2);
    _Float16* pWb1 = (_Float16*)carve((size_t)4 * 4 * 512 * 2);
    _Float16* pWa2 = (_Float16*)carve((size_t)5 * 4 * 512 * 2);
    _Float16* pWb2 = (_Float16*)carve((size_t)2 * 2 * 512 * 2);
    _Float16* pWa3 = (_Float16*)carve((size_t)3 * 2 * 512 * 2);
    _Float16* pWb3 = (_Float16*)carve((size_t)1 * 1 * 512 * 2);
    (void)ws_size; (void)n_in; (void)out_size;

    auto zblk = [](int n) { return (n + 255) / 256; };

    // ---- zero accumulators that persist across the whole pass ----
    zero_f32<<<zblk(N), 256, 0, stream>>>(cntN, N);
    zero_f32<<<zblk(G), 256, 0, stream>>>(cntG, G);
    zero_f32<<<zblk(G * 16), 256, 0, stream>>>(graphSum, G * 16);

    // ---- pack weights into WMMA B-fragment order ----
    pack_frags<<<zblk(5 * 8 * 512), 256, 0, stream>>>(W1a, pWa1, 5, 8, 128, 64);
    pack_frags<<<zblk(4 * 4 * 512), 256, 0, stream>>>(W1b, pWb1, 4, 4, 64, 0);
    pack_frags<<<zblk(5 * 4 * 512), 256, 0, stream>>>(W2a, pWa2, 5, 4, 64, 64);
    pack_frags<<<zblk(2 * 2 * 512), 256, 0, stream>>>(W2b, pWb2, 2, 2, 32, 0);
    pack_frags<<<zblk(3 * 2 * 512), 256, 0, stream>>>(W3a, pWa3, 3, 2, 32, 32);
    pack_frags<<<zblk(1 * 1 * 512), 256, 0, stream>>>(W3b, pWb3, 1, 1, 16, 0);

    // ---- embeddings & counts ----
    node_embed<<<N, 64, 0, stream>>>(x, Wn, bn_, hF16, N);
    edge_embed<<<zblk(E * 16), 256, 0, stream>>>(edge_attr, We, be_, eF16, E);
    edge_counts<<<zblk(E), 256, 0, stream>>>(dstIdx, cntN, E);
    node_counts<<<zblk(N), 256, 0, stream>>>(batch, cntG, N);

    // 4 waves/block, 4 tiles/wave -> 16 tiles per block
    const int nTiles = (E + 15) / 16;
    const int convBlocks = (nTiles + 16 - 1) / 16;
    const float invN = 1.0f / (float)N;

    // dynamic LDS sizes: Wa frags + Wb frags + 4 hidden tiles (16 x H f16)
    const size_t shm1 = (size_t)(5 * 8 + 4 * 4) * 1024 + 4 * 16 * 128 * 2; // 73728
    const size_t shm2 = (size_t)(5 * 4 + 2 * 2) * 1024 + 4 * 16 * 64 * 2;  // 32768
    const size_t shm3 = (size_t)(3 * 2 + 1 * 1) * 1024 + 4 * 16 * 32 * 2;  // 11264

    // ---- layer 1: 64 -> (144->128->64) ----
    zero_f32<<<zblk(N * 64), 256, 0, stream>>>(nodeSum, N * 64);
    conv_kernel<64, 128, 64><<<convBlocks, 128, shm1, stream>>>(
        hF16, eF16, srcIdx, dstIdx, pWa1, b1a, pWb1, b1b, nodeSum, E);
    zero_f32<<<1, 128, 0, stream>>>(stats, 128);
    finalize_stats<64><<<1024, 256, 0, stream>>>(nodeSum, cntN, stats, N);
    bn_params<<<1, 64, 0, stream>>>(stats, g1, be1, params, 64, invN);
    apply_bn<64, false><<<zblk(N * 64), 256, 0, stream>>>(nodeSum, params, hF16, batch, graphSum, N);

    // ---- layer 2: 64 -> (144->64->32) ----
    zero_f32<<<zblk(N * 32), 256, 0, stream>>>(nodeSum, N * 32);
    conv_kernel<64, 64, 32><<<convBlocks, 128, shm2, stream>>>(
        hF16, eF16, srcIdx, dstIdx, pWa2, b2a, pWb2, b2b, nodeSum, E);
    zero_f32<<<1, 128, 0, stream>>>(stats, 128);
    finalize_stats<32><<<1024, 256, 0, stream>>>(nodeSum, cntN, stats, N);
    bn_params<<<1, 32, 0, stream>>>(stats, g2, be2, params, 32, invN);
    apply_bn<32, false><<<zblk(N * 32), 256, 0, stream>>>(nodeSum, params, hF16, batch, graphSum, N);

    // ---- layer 3: 32 -> (80->32->16), fused graph pooling ----
    zero_f32<<<zblk(N * 16), 256, 0, stream>>>(nodeSum, N * 16);
    conv_kernel<32, 32, 16><<<convBlocks, 128, shm3, stream>>>(
        hF16, eF16, srcIdx, dstIdx, pWa3, b3a, pWb3, b3b, nodeSum, E);
    zero_f32<<<1, 128, 0, stream>>>(stats, 128);
    finalize_stats<16><<<1024, 256, 0, stream>>>(nodeSum, cntN, stats, N);
    bn_params<<<1, 16, 0, stream>>>(stats, g3, be3, params, 16, invN);
    apply_bn<16, true><<<zblk(N * 16), 256, 0, stream>>>(nodeSum, params, hF16, batch, graphSum, N);

    // ---- head ----
    head<<<zblk(G * 12), 256, 0, stream>>>(graphSum, cntG, Wfc, bfc, (float*)d_out, G);
}
